// RGCNEncoder_7619271983570
// MI455X (gfx1250) — matrix operand order repaired and checked by reading, not compiled, not run on hardware
//
#include <hip/hip_runtime.h>
#include <hip/hip_bf16.h>
#include <cstdint>

// ---------------------------------------------------------------------------
// CDNA5 (gfx1250) RGCN encoder forward.
//  - GEMMs:  v_wmma_f32_16x16x32_bf16 (bf16 A/B, fp32 acc), async-to-LDS
//            staged A tile (GLOBAL_LOAD_ASYNC_TO_LDS_B128 / ASYNCcnt),
//            2 accumulators/wave, branch-free epilogue.
//  - Edges:  blockIdx-uniform index loads + global_atomic_add_f32 scatter
//  - LN:     wave32 __shfl_xor reductions
// ---------------------------------------------------------------------------

typedef __attribute__((ext_vector_type(16))) __bf16 v16bf;
typedef __attribute__((ext_vector_type(8)))  __bf16 v8bf;
typedef __attribute__((ext_vector_type(8)))  float  v8f;
typedef __attribute__((ext_vector_type(4)))  int    v4i;

#define HID_K 128  // inner dimension of every GEMM in this model

#if __has_builtin(__builtin_amdgcn_global_load_async_to_lds_b128)
#define ASYNC_LDS_COPY 1
typedef __attribute__((address_space(1))) v4i* gv4p;  // global (AS1) int4*
typedef __attribute__((address_space(3))) v4i* lv4p;  // LDS    (AS3) int4*
#else
#define ASYNC_LDS_COPY 0
#endif

// ---------------- elementwise / prep kernels ----------------

__global__ void k_zero_f32(float* __restrict__ p, int n) {
  int i = blockIdx.x * blockDim.x + threadIdx.x;
  if (i < n) p[i] = 0.0f;
}

__global__ void k_cvt_f32_bf16(const float* __restrict__ in, __bf16* __restrict__ out, int n) {
  int i = blockIdx.x * blockDim.x + threadIdx.x;
  if (i < n) out[i] = (__bf16)in[i];
}

// W [Kd x N] f32 row-major  ->  Wt [N x Kd] bf16 (so B columns are contiguous)
__global__ void k_transpose_bf16(const float* __restrict__ W, __bf16* __restrict__ Wt,
                                 int Kd, int N) {
  int i = blockIdx.x * blockDim.x + threadIdx.x;
  if (i >= Kd * N) return;
  int k = i / N, n = i - k * N;
  Wt[(size_t)n * Kd + k] = (__bf16)W[i];
}

// agg[r][c] = ba[c] + bb[c]   (sum of the two relation biases targeting this type)
__global__ void k_init_bias2(float* __restrict__ agg, const float* __restrict__ ba,
                             const float* __restrict__ bb, int rows, int C) {
  int i = blockIdx.x * blockDim.x + threadIdx.x;
  if (i >= rows * C) return;
  int c = i % C;
  agg[i] = ba[c] + bb[c];
}

// ---------------- degree / normalization ----------------

__global__ void k_degree(const int* __restrict__ src, const int* __restrict__ dst,
                         float* __restrict__ ds, float* __restrict__ dd, int E) {
  int e = blockIdx.x * blockDim.x + threadIdx.x;
  if (e >= E) return;
  atomicAdd(ds + src[e], 1.0f);
  atomicAdd(dd + dst[e], 1.0f);
}

__global__ void k_inv_sqrt(float* __restrict__ d, int n) {
  int i = blockIdx.x * blockDim.x + threadIdx.x;
  if (i >= n) return;
  float v = d[i];
  d[i] = (v > 0.0f) ? rsqrtf(v) : 0.0f;
}

// ---------------- WMMA GEMM:  C[M,N] = A[M,128] @ Wt[N,128]^T (+bias) -------
// Block tile: 32 rows x N cols. Waves = 2 m-subtiles x (N/32) n-pairs; each
// wave owns two 16x16 output tiles (two accumulators) so every A fragment
// feeds two v_wmma instructions. The 32x128 bf16 A tile (8 KB) is staged in
// LDS via GLOBAL_LOAD_ASYNC_TO_LDS_B128 (no VGPR round-trip, ASYNCcnt),
// falling back to a coalesced b128 copy. Requires M % 32 == 0 (20000/60000 ok).
// A frag (16-bit A 16x32): lane m: K 0-15 ; lane m+16: K 16-31 (matches B).
// B frag (16-bit B 32x16): lane n: K 0-15 ; lane n+16: K 16-31 (contig in Wt).
// C/D: lane = column n, acc[j] = row (j + 8*(lane>=16)).
template <int N, bool HASB, bool OUTF, bool OUTB>
__global__ __launch_bounds__(2 * (N / 32) * 32)
void k_gemm_wmma(const __bf16* __restrict__ A,    // [M x 128] bf16 row-major
                 const __bf16* __restrict__ Wt,   // [N x 128] bf16 row-major
                 const float* __restrict__ bias,  // [N] (read iff HASB)
                 float* __restrict__ Cf,          // [M x N] f32  (written iff OUTF)
                 __bf16* __restrict__ Cb)         // [M x N] bf16 (written iff OUTB)
{
  constexpr int NWAVES = 2 * (N / 32);
  constexpr int TPB    = NWAVES * 32;
  __shared__ __attribute__((aligned(16))) __bf16 As[32 * HID_K];  // 8 KB

  const int tid  = threadIdx.x;
  const int lane = tid & 31;
  const int wave = tid >> 5;
  const int m0   = blockIdx.x * 32;

  // ---- stage the 32x128 A tile into LDS ----
  constexpr int NCHUNK = 32 * HID_K / 8;  // 512 x (8 bf16 = 16B)
  const __bf16* gA = A + (size_t)m0 * HID_K;
#if ASYNC_LDS_COPY
#pragma unroll
  for (int c = tid; c < NCHUNK; c += TPB)
    __builtin_amdgcn_global_load_async_to_lds_b128(
        (gv4p)(gA + c * 8), (lv4p)(As + c * 8), 0, 0);
#if __has_builtin(__builtin_amdgcn_s_wait_asynccnt)
  __builtin_amdgcn_s_wait_asynccnt(0);
#else
  asm volatile("s_wait_asynccnt 0" ::: "memory");
#endif
#else
#pragma unroll
  for (int c = tid; c < NCHUNK; c += TPB)
    *(v8bf*)(As + c * 8) = *(const v8bf*)(gA + c * 8);
#endif
  __syncthreads();

  const int waveM = wave & 1;   // m-subtile (0/1)
  const int waveN = wave >> 1;  // n-pair index
  const int n0    = waveN * 32;

  const int half  = lane >> 4;  // 0 or 1
  const int l15   = lane & 15;
  const int akoff = half << 4;  // A: 0 or 16
  const int bkoff = half << 4;  // B: 0 or 16

  const __bf16* arow  = As + (waveM * 16 + l15) * HID_K;
  const __bf16* brow0 = Wt + (size_t)(n0 + l15) * HID_K;
  const __bf16* brow1 = brow0 + 16 * HID_K;

  v8f acc0 = {}, acc1 = {};
#pragma unroll
  for (int k0 = 0; k0 < HID_K; k0 += 32) {
    const v8bf alo = *(const v8bf*)(arow + k0 + akoff);        // ds_load_b128
    const v8bf ahi = *(const v8bf*)(arow + k0 + akoff + 8);    // ds_load_b128
    v16bf afrag;
#pragma unroll
    for (int j = 0; j < 8; ++j) { afrag[j] = alo[j]; afrag[8 + j] = ahi[j]; }
    const v16bf b0 = *(const v16bf*)(brow0 + k0 + bkoff);
    const v16bf b1 = *(const v16bf*)(brow1 + k0 + bkoff);
    acc0 = __builtin_amdgcn_wmma_f32_16x16x32_bf16(false, afrag, false, b0,
                                                   (short)0, acc0, false, false);
    acc1 = __builtin_amdgcn_wmma_f32_16x16x32_bf16(false, afrag, false, b1,
                                                   (short)0, acc1, false, false);
  }

  // ---- branch-free epilogue ----
  const int   mbase = m0 + waveM * 16 + (half << 3);
  const int   nA = n0 + l15, nB = n0 + 16 + l15;
  const float bv0 = HASB ? bias[nA] : 0.0f;
  const float bv1 = HASB ? bias[nB] : 0.0f;
#pragma unroll
  for (int j = 0; j < 8; ++j) {
    const size_t r  = (size_t)(mbase + j) * N;
    const float  y0 = acc0[j] + bv0;
    const float  y1 = acc1[j] + bv1;
    if constexpr (OUTF) { Cf[r + nA] = y0; Cf[r + nB] = y1; }
    if constexpr (OUTB) { Cb[r + nA] = (__bf16)y0; Cb[r + nB] = (__bf16)y1; }
  }
}

// ---------------- edge scatter:  agg[dst] += t[src] * inv_s[src]*inv_d[dst] -
// EPB edges per block, processed with a uniform unrolled loop so src/dst/norm
// stay blockIdx-uniform (scalar loads). One lane per channel -> coalesced
// gather + global_atomic_add_f32 into the L2-resident aggregate.
template <int C, int EPB>
__global__ void k_scatter(const float* __restrict__ t, const int* __restrict__ src,
                          const int* __restrict__ dst, const float* __restrict__ invs,
                          const float* __restrict__ invd, float* __restrict__ agg, int E) {
  const int c = threadIdx.x;
  const int ebase = blockIdx.x * EPB;
#pragma unroll
  for (int k = 0; k < EPB; ++k) {
    const int e = ebase + k;
    if (e >= E) return;  // uniform
    const int s = src[e];
    const int d = dst[e];
    const float norm = invs[s] * invd[d];
    atomicAdd(agg + (size_t)d * C + c, t[(size_t)s * C + c] * norm);
  }
}

// ---------------- wave-per-row LayerNorm (+optional ReLU) -------------------
template <int C, bool OUTF, bool OUTB>
__global__ void k_layernorm(const float* __restrict__ in, const float* __restrict__ g,
                            const float* __restrict__ b, float* __restrict__ outf,
                            __bf16* __restrict__ outb, int n, int relu) {
  const int row  = blockIdx.x * (blockDim.x >> 5) + (threadIdx.x >> 5);
  const int lane = threadIdx.x & 31;
  if (row >= n) return;
  const float* xr = in + (size_t)row * C;
  constexpr int PER = C / 32;
  float v[PER];
  float s = 0.0f;
#pragma unroll
  for (int i = 0; i < PER; ++i) { v[i] = xr[lane + (i << 5)]; s += v[i]; }
#pragma unroll
  for (int o = 16; o > 0; o >>= 1) s += __shfl_xor(s, o, 32);
  const float mu = s * (1.0f / C);
  float q = 0.0f;
#pragma unroll
  for (int i = 0; i < PER; ++i) { float dd = v[i] - mu; q += dd * dd; }
#pragma unroll
  for (int o = 16; o > 0; o >>= 1) q += __shfl_xor(q, o, 32);
  const float rstd = rsqrtf(q * (1.0f / C) + 1e-5f);
#pragma unroll
  for (int i = 0; i < PER; ++i) {
    const int c = lane + (i << 5);
    float y = (v[i] - mu) * rstd * g[c] + b[c];
    if (relu) y = fmaxf(y, 0.0f);
    if constexpr (OUTF) outf[(size_t)row * C + c] = y;
    if constexpr (OUTB) outb[(size_t)row * C + c] = (__bf16)y;
  }
}

// ---------------------------------------------------------------------------
// Host orchestration (graph-capture safe: only kernel launches on `stream`)
// ---------------------------------------------------------------------------
extern "C" void kernel_launch(void* const* d_in, const int* in_sizes, int n_in,
                              void* d_out, int out_size, void* d_ws, size_t ws_size,
                              hipStream_t stream) {
  (void)n_in; (void)out_size; (void)ws_size;

  const float* x[3]  = {(const float*)d_in[0], (const float*)d_in[1], (const float*)d_in[2]};
  const float* Wp[3] = {(const float*)d_in[3], (const float*)d_in[5], (const float*)d_in[7]};
  const float* bp[3] = {(const float*)d_in[4], (const float*)d_in[6], (const float*)d_in[8]};
  const float* W1    = (const float*)d_in[9];
  const float* b1    = (const float*)d_in[10];
  const float* W2    = (const float*)d_in[11];
  const float* b2    = (const float*)d_in[12];
  const float* ln1_g = (const float*)d_in[13];
  const float* ln1_b = (const float*)d_in[14];
  const float* ln2_g = (const float*)d_in[15];
  const float* ln2_b = (const float*)d_in[16];

  const int    SZ[3]   = {20000, 60000, 20000};     // drug, gene, disease
  const size_t roff[3] = {0, 20000, 80000};         // row offsets in concatenated node order
  const int    es[6]   = {0, 0, 1, 2, 1, 2};        // relation src types
  const int    ed[6]   = {2, 1, 2, 0, 0, 1};        // relation dst types
  const int    dstrel[3][2] = {{3, 4}, {1, 5}, {0, 2}};  // relations targeting each type
  const int    E = in_sizes[17] / 2;
  const int    EPB = 4;                              // edges per scatter block

  float* out = (float*)d_out;

  // ---- workspace carve ----
  uintptr_t cur = (uintptr_t)d_ws;
  auto carve = [&](size_t bytes) -> void* {
    void* p = (void*)cur;
    cur += (bytes + 255) & ~(size_t)255;
    return p;
  };
  const size_t NTOT = 100000;
  float*  agg1 = (float*) carve(NTOT * 128 * sizeof(float));          // 51.2 MB (L2-resident)
  float*  tbuf = (float*) carve((size_t)60000 * 128 * sizeof(float)); // 30.7 MB
  __bf16* h0b  = (__bf16*)carve(NTOT * 128 * sizeof(__bf16));         // 25.6 MB
  __bf16* h1b  = (__bf16*)carve(NTOT * 128 * sizeof(__bf16));         // 25.6 MB
  __bf16* Wpt  = (__bf16*)carve((size_t)3 * 128 * 128 * sizeof(__bf16));
  __bf16* W1t  = (__bf16*)carve((size_t)6 * 128 * 128 * sizeof(__bf16));
  __bf16* W2t  = (__bf16*)carve((size_t)6 * 64 * 128 * sizeof(__bf16));
  float* invS[6]; float* invD[6];
  for (int i = 0; i < 6; ++i) {
    invS[i] = (float*)carve((size_t)SZ[es[i]] * sizeof(float));
    invD[i] = (float*)carve((size_t)SZ[ed[i]] * sizeof(float));
  }
  __bf16* xb   = (__bf16*)agg1;  // transient alias: dead before agg1 is initialized
  float*  agg2 = agg1;           // layer-2 aggregate reuses agg1 space (after LN1)

  const int T = 256;
  auto cdiv = [](int a, int b) { return (a + b - 1) / b; };

  // ---- weights -> transposed bf16 ----
  for (int t = 0; t < 3; ++t)
    k_transpose_bf16<<<cdiv(128 * 128, T), T, 0, stream>>>(
        Wp[t], Wpt + (size_t)t * 128 * 128, 128, 128);
  for (int i = 0; i < 6; ++i)
    k_transpose_bf16<<<cdiv(128 * 128, T), T, 0, stream>>>(
        W1 + (size_t)i * 128 * 128, W1t + (size_t)i * 128 * 128, 128, 128);
  for (int i = 0; i < 6; ++i)
    k_transpose_bf16<<<cdiv(128 * 64, T), T, 0, stream>>>(
        W2 + (size_t)i * 128 * 64, W2t + (size_t)i * 64 * 128, 128, 64);

  // ---- per-relation degree normalization factors ----
  for (int i = 0; i < 6; ++i) {
    const int ns = SZ[es[i]], nd = SZ[ed[i]];
    const int* ei = (const int*)d_in[17 + i];
    k_zero_f32<<<cdiv(ns, T), T, 0, stream>>>(invS[i], ns);
    k_zero_f32<<<cdiv(nd, T), T, 0, stream>>>(invD[i], nd);
    k_degree<<<cdiv(E, T), T, 0, stream>>>(ei, ei + E, invS[i], invD[i], E);
    k_inv_sqrt<<<cdiv(ns, T), T, 0, stream>>>(invS[i], ns);
    k_inv_sqrt<<<cdiv(nd, T), T, 0, stream>>>(invD[i], nd);
  }

  // ---- input projection: h0 = bf16(x @ Wp + bp) ----
  for (int t = 0; t < 3; ++t)
    k_cvt_f32_bf16<<<cdiv(SZ[t] * 128, T), T, 0, stream>>>(
        x[t], xb + roff[t] * 128, SZ[t] * 128);
  for (int t = 0; t < 3; ++t)
    k_gemm_wmma<128, true, false, true><<<SZ[t] / 32, 256, 0, stream>>>(
        xb + roff[t] * 128, Wpt + (size_t)t * 128 * 128, bp[t],
        nullptr, h0b + roff[t] * 128);

  // ---- layer 1: HeteroConv(sum) of GCNConv(128->128), then LN + ReLU ----
  for (int t = 0; t < 3; ++t)
    k_init_bias2<<<cdiv(SZ[t] * 128, T), T, 0, stream>>>(
        agg1 + roff[t] * 128, b1 + (size_t)dstrel[t][0] * 128,
        b1 + (size_t)dstrel[t][1] * 128, SZ[t], 128);
  for (int i = 0; i < 6; ++i) {
    const int s = es[i], d = ed[i];
    const int* ei = (const int*)d_in[17 + i];
    k_gemm_wmma<128, false, true, false><<<SZ[s] / 32, 256, 0, stream>>>(
        h0b + roff[s] * 128, W1t + (size_t)i * 128 * 128, nullptr, tbuf, nullptr);
    k_scatter<128, 4><<<cdiv(E, EPB), 128, 0, stream>>>(
        tbuf, ei, ei + E, invS[i], invD[i], agg1 + roff[d] * 128, E);
  }
  for (int t = 0; t < 3; ++t)
    k_layernorm<128, false, true><<<cdiv(SZ[t], 8), 256, 0, stream>>>(
        agg1 + roff[t] * 128, ln1_g + (size_t)t * 128, ln1_b + (size_t)t * 128,
        nullptr, h1b + roff[t] * 128, SZ[t], /*relu=*/1);

  // ---- layer 2: GCNConv(128->64), then LN -> d_out ----
  for (int t = 0; t < 3; ++t)
    k_init_bias2<<<cdiv(SZ[t] * 64, T), T, 0, stream>>>(
        agg2 + roff[t] * 64, b2 + (size_t)dstrel[t][0] * 64,
        b2 + (size_t)dstrel[t][1] * 64, SZ[t], 64);
  for (int i = 0; i < 6; ++i) {
    const int s = es[i], d = ed[i];
    const int* ei = (const int*)d_in[17 + i];
    k_gemm_wmma<64, false, true, false><<<SZ[s] / 32, 128, 0, stream>>>(
        h1b + roff[s] * 128, W2t + (size_t)i * 64 * 128, nullptr, tbuf, nullptr);
    k_scatter<64, 4><<<cdiv(E, EPB), 64, 0, stream>>>(
        tbuf, ei, ei + E, invS[i], invD[i], agg2 + roff[d] * 64, E);
  }
  for (int t = 0; t < 3; ++t)
    k_layernorm<64, true, false><<<cdiv(SZ[t], 8), 256, 0, stream>>>(
        agg2 + roff[t] * 64, ln2_g + (size_t)t * 64, ln2_b + (size_t)t * 64,
        out + roff[t] * 64, nullptr, SZ[t], /*relu=*/0);
}
